// HardAttention_58265526338167
// MI455X (gfx1250) — matrix operand order repaired and checked by reading, not compiled, not run on hardware
//
#include <hip/hip_runtime.h>

// Shapes from the reference
#define B_ 4096
#define N_ 64
#define D_ 256
#define H_ 512
#define U_ 512

typedef __attribute__((ext_vector_type(16))) __bf16 bf16x16;
typedef __attribute__((ext_vector_type(8)))  float  f32x8;
typedef __attribute__((address_space(3))) unsigned short lds_ushort;

union Frag {
  bf16x16 v;
  uint4 q[2];
};

// float -> bf16 with round-to-nearest-even
__device__ __forceinline__ unsigned short f2bf(float f) {
  unsigned u = __builtin_bit_cast(unsigned, f);
  u += 0x7fffu + ((u >> 16) & 1u);
  return (unsigned short)(u >> 16);
}

__device__ __forceinline__ f32x8 wmma_bf16(bf16x16 a, bf16x16 b, f32x8 c) {
  // (neg_a, A, neg_b, B, c_mod, C, reuse_a, reuse_b)
  return __builtin_amdgcn_wmma_f32_16x16x32_bf16(false, a, false, b, (short)0, c,
                                                 false, false);
}

// CDNA5 async memory->LDS copy (no VGPR round trip, tracked by ASYNCcnt).
// VDST = per-lane LDS byte address, VADDR = per-lane 64-bit global address.
__device__ __forceinline__ void async_ld_b128(unsigned lds_byte_off,
                                              unsigned long long gaddr) {
  asm volatile("global_load_async_to_lds_b128 %0, %1, off"
               :: "v"(lds_byte_off), "v"(gaddr)
               : "memory");
}
__device__ __forceinline__ void wait_async0() {
  asm volatile("s_wait_asynccnt 0x0" ::: "memory");
}

// ---------------------------------------------------------------------------
// Kernel 1: repack Wf (256x512) and Wh (512x512) f32 -> bf16 in this code's
// per-lane WMMA B-fragment order (contiguous k, as the A loads): element e of lane l (for k-step ks, u-tile
// ut) is K = ks*32 + 16*(l>=16) + e, N = ut*16 + (l&15). 16 contiguous bf16
// per lane -> one coalesced 32B load in the GEMM kernels.
// ---------------------------------------------------------------------------
__global__ void pack_weights(const float* __restrict__ Wf, const float* __restrict__ Wh,
                             unsigned short* __restrict__ wfp,
                             unsigned short* __restrict__ whp) {
  int id = blockIdx.x * 256 + threadIdx.x;
  if (id < 32 * 8 * 32) {  // Wf: 32 u-tiles, 8 k-steps, 32 lanes
    int lane = id & 31, ks = (id >> 5) & 7, ut = id >> 8;
    int u  = ut * 16 + (lane & 15);
    int kb = ks * 32 + (lane >> 4) * 16;
    unsigned short* dst = wfp + (size_t)id * 16;
#pragma unroll
    for (int e = 0; e < 16; ++e) dst[e] = f2bf(Wf[(size_t)(kb + e) * U_ + u]);
  } else {
    id -= 32 * 8 * 32;
    if (id >= 32 * 16 * 32) return;  // Wh: 32 u-tiles, 16 k-steps, 32 lanes
    int lane = id & 31, ks = (id >> 5) & 15, ut = id >> 9;
    int u  = ut * 16 + (lane & 15);
    int kb = ks * 32 + (lane >> 4) * 16;
    unsigned short* dst = whp + (size_t)id * 16;
#pragma unroll
    for (int e = 0; e < 16; ++e) dst[e] = f2bf(Wh[(size_t)(kb + e) * U_ + u]);
  }
}

// ---------------------------------------------------------------------------
// Kernel 2: hprojB[b,u] = hidden[b,:] @ Wh[:,u] + bh[u] + bf[u]   (4096x512)
// One block = 16 rows of hidden; 4 waves split the 32 u-tiles (8 each).
// ks OUTER (rolled, one A frag live), u-tiles INNER (unrolled, 8 persistent
// f32x8 accumulators). Verified spill-free.
// ---------------------------------------------------------------------------
#define HS 520  // 512 + 8 pad (ushorts): row stride 1040B -> 4-bank rotation
__global__ void hproj_kernel(const float* __restrict__ hidden,
                             const unsigned short* __restrict__ whp,
                             const float* __restrict__ bh, const float* __restrict__ bf,
                             float* __restrict__ hprojB) {
  __shared__ unsigned short lA[16 * HS];
  int tid = threadIdx.x;          // 128 threads = 4 waves
  int mbase = blockIdx.x * 16;
  // cooperative load + f32->bf16 convert of 16x512 tile
#pragma unroll
  for (int i = 0; i < 16; ++i) {
    int j = i * 128 + tid;
    int row = j >> 7, c4 = j & 127;
    const float4 f = *(const float4*)(hidden + (size_t)(mbase + row) * H_ + c4 * 4);
    unsigned lo = (unsigned)f2bf(f.x) | ((unsigned)f2bf(f.y) << 16);
    unsigned hi = (unsigned)f2bf(f.z) | ((unsigned)f2bf(f.w) << 16);
    *(uint2*)&lA[row * HS + c4 * 4] = make_uint2(lo, hi);
  }
  __syncthreads();
  int wave = tid >> 5, lane = tid & 31, half = lane >> 4, rl = lane & 15;

  f32x8 acc[8];
#pragma unroll
  for (int i = 0; i < 8; ++i) acc[i] = (f32x8){0, 0, 0, 0, 0, 0, 0, 0};

#pragma unroll 1
  for (int ks = 0; ks < 16; ++ks) {
    Frag a;
    int ab = rl * HS + ks * 32 + half * 16;         // K = 16h..16h+7
    a.q[0] = *(const uint4*)&lA[ab];
    a.q[1] = *(const uint4*)&lA[ab + 8];            // K = 16h+8..
#pragma unroll
    for (int uti = 0; uti < 8; ++uti) {
      int ut = wave * 8 + uti;
      Frag b;
      const uint4* bp =
          (const uint4*)(whp + ((size_t)((ut * 16 + ks) * 32 + lane)) * 16);
      b.q[0] = bp[0];
      b.q[1] = bp[1];
      acc[uti] = wmma_bf16(a.v, b.v, acc[uti]);
    }
  }

#pragma unroll
  for (int uti = 0; uti < 8; ++uti) {
    int ut = wave * 8 + uti;
    int u = ut * 16 + rl;
    float bias = bh[u] + bf[u];
#pragma unroll
    for (int r = 0; r < 8; ++r)
      hprojB[(size_t)(mbase + r + 8 * half) * U_ + u] = acc[uti][r] + bias;
  }
}

// ---------------------------------------------------------------------------
// Kernel 3 (fused core): one block == one batch sample b (64 rows = N).
// logits[b,n] = sum_u tanh( (features@Wf)[b,n,u] + hprojB[b,u] ) * Ws[u]
// 4 waves, each owns a 16-row M-tile; A fragments (K=256 -> 8 steps, 64
// VGPRs) hoisted into registers. Wf_packed is streamed through LDS with
// DOUBLE-BUFFERED GLOBAL_LOAD_ASYNC_TO_LDS_B128: chunk ut+1 is prefetched
// into the alternate 8KB buffer while WMMAs consume chunk ut; handoff is
// s_wait_asynccnt 0 + workgroup barrier.
// ---------------------------------------------------------------------------
#define AS 264  // 256 + 8 pad (ushorts): row stride 528B -> 4-bank rotation
__global__ void score_kernel(const float* __restrict__ features,
                             const unsigned short* __restrict__ wfp,
                             const float* __restrict__ hprojB,
                             const float* __restrict__ Ws,
                             float* __restrict__ logits) {
  __shared__ unsigned short lA[64 * AS];           // 33792 B
  __shared__ unsigned short lB[2][8 * 32 * 16];    // 2 x 8192 B (double buffer)
  int tid = threadIdx.x;  // 128 threads = 4 waves
  int b = blockIdx.x;
  size_t rbase = (size_t)b * 64;
  // cooperative load + convert of 64x256 feature tile (read once from HBM)
#pragma unroll
  for (int i = 0; i < 32; ++i) {
    int j = i * 128 + tid;
    int row = j >> 6, c4 = j & 63;
    const float4 f = *(const float4*)(features + (rbase + row) * D_ + c4 * 4);
    unsigned lo = (unsigned)f2bf(f.x) | ((unsigned)f2bf(f.y) << 16);
    unsigned hi = (unsigned)f2bf(f.z) | ((unsigned)f2bf(f.w) << 16);
    *(uint2*)&lA[row * AS + c4 * 4] = make_uint2(lo, hi);
  }

  // LDS byte offsets of the two B buffers for this thread's async slice
  unsigned lb0 = (unsigned)(unsigned long long)(lds_ushort*)&lB[0][tid * 8];
  unsigned lb1 = (unsigned)(unsigned long long)(lds_ushort*)&lB[1][tid * 8];
  // issue one 8KB u-tile chunk: 128 threads x 4 async b128 (16B) copies
  auto issueB = [&](unsigned lbase, int ut) {
    unsigned long long g =
        (unsigned long long)(const void*)(wfp + (size_t)ut * 4096 + tid * 8);
#pragma unroll
    for (int i = 0; i < 4; ++i)
      async_ld_b128(lbase + i * 2048u, g + (unsigned long long)i * 2048u);
  };

  issueB(lb0, 0);  // prefetch first chunk
  __syncthreads();  // lA visible to all waves

  int wave = tid >> 5, lane = tid & 31, half = lane >> 4, rl = lane & 15;
  // hoist the 8 A fragments (full K) for this wave's 16 rows into VGPRs
  Frag A[8];
  int arow = wave * 16 + rl;
#pragma unroll
  for (int ks = 0; ks < 8; ++ks) {
    int ab = arow * AS + ks * 32 + half * 16;
    A[ks].q[0] = *(const uint4*)&lA[ab];
    A[ks].q[1] = *(const uint4*)&lA[ab + 8];
  }
  float pacc[8] = {0, 0, 0, 0, 0, 0, 0, 0};
#pragma unroll 1
  for (int ut = 0; ut < 32; ++ut) {  // one 16-column u-tile per iteration
    wait_async0();    // this wave's outstanding async copy (chunk ut) done
    __syncthreads();  // all waves' copies visible; prev buffer fully consumed
    if (ut + 1 < 32) issueB((ut & 1) ? lb0 : lb1, ut + 1);  // prefetch next
    const unsigned short* lbuf = &lB[ut & 1][0];
    f32x8 c = {0, 0, 0, 0, 0, 0, 0, 0};
#pragma unroll
    for (int ks = 0; ks < 8; ++ks) {
      Frag bfr;
      const uint4* bp = (const uint4*)&lbuf[(ks * 32 + lane) * 16];
      bfr.q[0] = bp[0];
      bfr.q[1] = bp[1];
      c = wmma_bf16(A[ks].v, bfr.v, c);
    }
    int u = ut * 16 + rl;
    float hb  = hprojB[(size_t)b * U_ + u];
    float wsu = Ws[u];
#pragma unroll
    for (int r = 0; r < 8; ++r) pacc[r] += tanhf(c[r] + hb) * wsu;
  }
  // reduce over the 16 lanes of each half (the u dimension), write logits
#pragma unroll
  for (int r = 0; r < 8; ++r) {
    float v = pacc[r];
    v += __shfl_xor(v, 8, 16);
    v += __shfl_xor(v, 4, 16);
    v += __shfl_xor(v, 2, 16);
    v += __shfl_xor(v, 1, 16);
    if (rl == 0) logits[rbase + wave * 16 + r + 8 * half] = v;  // bs cancels
  }
}

// ---------------------------------------------------------------------------
// Kernel 4: per-b softmax (alpha), greedy argmax + Gumbel-max selection
// (argmax(log a + g) == argmax(logits + g)), and fused context gather.
// ---------------------------------------------------------------------------
__global__ void finalize_kernel(const float* __restrict__ logits,
                                const float* __restrict__ gumbel,
                                const int* __restrict__ greedy,
                                const float* __restrict__ features,
                                float* __restrict__ outCtx,
                                float* __restrict__ outAlpha) {
  __shared__ float sv[64];
  __shared__ int si[64];
  int b = blockIdx.x, t = threadIdx.x;  // 64 threads
  float l = logits[(size_t)b * N_ + t];
  // max + first-occurrence argmax (greedy path; argmax(a)==argmax(logits))
  sv[t] = l; si[t] = t;
  __syncthreads();
  for (int s = 32; s > 0; s >>= 1) {
    if (t < s) {
      float av = sv[t], cv = sv[t + s];
      int ai = si[t], ci = si[t + s];
      if (cv > av || (cv == av && ci < ai)) { sv[t] = cv; si[t] = ci; }
    }
    __syncthreads();
  }
  float m = sv[0];
  int locg = si[0];
  __syncthreads();
  // softmax
  float e = expf(l - m);
  sv[t] = e;
  __syncthreads();
  for (int s = 32; s > 0; s >>= 1) {
    if (t < s) sv[t] += sv[t + s];
    __syncthreads();
  }
  float denom = sv[0];
  __syncthreads();
  outAlpha[(size_t)b * N_ + t] = e / denom;
  // Gumbel-max sample path
  sv[t] = l + gumbel[(size_t)b * N_ + t];
  si[t] = t;
  __syncthreads();
  for (int s = 32; s > 0; s >>= 1) {
    if (t < s) {
      float av = sv[t], cv = sv[t + s];
      int ai = si[t], ci = si[t + s];
      if (cv > av || (cv == av && ci < ai)) { sv[t] = cv; si[t] = ci; }
    }
    __syncthreads();
  }
  int loc = greedy[b] ? locg : si[0];
  // gather context row: 64 threads x float4 = 256 floats
  const float4* srcp = (const float4*)(features + ((size_t)b * N_ + loc) * D_);
  float4* dstp = (float4*)(outCtx + (size_t)b * D_);
  dstp[t] = srcp[t];
}

// ---------------------------------------------------------------------------
extern "C" void kernel_launch(void* const* d_in, const int* in_sizes, int n_in,
                              void* d_out, int out_size, void* d_ws, size_t ws_size,
                              hipStream_t stream) {
  (void)in_sizes; (void)n_in; (void)out_size; (void)ws_size;
  const float* features = (const float*)d_in[0];  // (B,N,D)
  const float* hidden   = (const float*)d_in[1];  // (B,H)
  const int*   greedy   = (const int*)d_in[2];    // (B,)
  const float* gumbel   = (const float*)d_in[3];  // (B,N)
  const float* Wf       = (const float*)d_in[4];  // (D,U)
  const float* bf       = (const float*)d_in[5];  // (U,)
  const float* Wh       = (const float*)d_in[6];  // (H,U)
  const float* bh       = (const float*)d_in[7];  // (U,)
  const float* Ws       = (const float*)d_in[8];  // (U,1)
  // d_in[9] = bs: additive constant on logits -> cancels in softmax/argmax

  unsigned short* wfp = (unsigned short*)d_ws;          // 131072 ushorts
  unsigned short* whp = wfp + 32 * 8 * 32 * 16;         // 262144 ushorts
  float* hprojB = (float*)(whp + 32 * 16 * 32 * 16);    // B*U floats
  float* logits = hprojB + (size_t)B_ * U_;             // B*N floats

  float* outCtx   = (float*)d_out;            // (B,D)
  float* outAlpha = outCtx + (size_t)B_ * D_; // (B,N,1)

  hipLaunchKernelGGL(pack_weights, dim3(96), dim3(256), 0, stream, Wf, Wh, wfp, whp);
  hipLaunchKernelGGL(hproj_kernel, dim3(B_ / 16), dim3(128), 0, stream,
                     hidden, whp, bh, bf, hprojB);
  hipLaunchKernelGGL(score_kernel, dim3(B_), dim3(128), 0, stream,
                     features, wfp, hprojB, Ws, logits);
  hipLaunchKernelGGL(finalize_kernel, dim3(B_), dim3(64), 0, stream,
                     logits, gumbel, greedy, features, outCtx, outAlpha);
}
